// UFNO_80032420594357
// MI455X (gfx1250) — compile-verified, matrix operand order checked by reading, not run on hardware
//
#include <hip/hip_runtime.h>
#include <hip/hip_bf16.h>
#include <stdint.h>

// ---------------------------------------------------------------------------
// UFNO forward for MI455X (gfx1250, wave32).
// All conv-like ops (3x3 conv via implicit im2col, 1x1 conv, conv-transpose)
// run through one WMMA GEMM kernel using v_wmma_f32_16x16x32_f16 (f16 in,
// f32 accumulate). A-panel staging uses async global->LDS (ASYNCcnt) when
// available; all loaders are branchless with shift/mask addressing.
// ---------------------------------------------------------------------------

typedef _Float16 half_t;
typedef __attribute__((ext_vector_type(16))) _Float16 v16h;
typedef __attribute__((ext_vector_type(8)))  _Float16 v8h;
typedef __attribute__((ext_vector_type(8)))  float    v8f;
typedef __attribute__((ext_vector_type(4)))  int      v4i;

#if __has_builtin(__builtin_amdgcn_global_load_async_to_lds_b128)
#define HAVE_ASYNC_LDS 1
#else
#define HAVE_ASYNC_LDS 0
#endif

static __device__ __forceinline__ void wait_async_lds0() {
#if __has_builtin(__builtin_amdgcn_s_wait_asynccnt)
  __builtin_amdgcn_s_wait_asynccnt(0);
#else
  asm volatile("s_wait_asynccnt 0x0" ::: "memory");
#endif
}

#define BM 32
#define BN 64
#define BK 64
#define KPAD 72  // LDS row pitch in halves (144B, multiple of 16B)

enum { BMODE_CONV1 = 0, BMODE_CONV3 = 1 };
enum { EP_PLAIN = 0, EP_UP2 = 1 };

static inline int ceil_div(int a, int b) { return (a + b - 1) / b; }
static inline int ilog2i(int v) { int l = 0; while ((1 << l) < v) ++l; return l; }

// ---------------------------------------------------------------------------
// WMMA GEMM:  out[M,N](f32) = A[M,K](f16) * B[K,N](f16, generated on the fly)
//   BMODE_CONV1: B[k,n] = act[b, k, hw]            (n = b*HW + hw)
//   BMODE_CONV3: B[k,n] = act[b, ci, y+dy, x+dx]   (k = ci*9 + (dy+1)*3+(dx+1))
//   EP_PLAIN:    out[(b*M+m)*HW + hw]              (+ bias[m])
//   EP_UP2:      out[((b*M+m)*2H + 2y+upD)*2W + 2x+upK]
// Spatial size is always square power-of-two: W == H == 1<<lgW.
// Block: 256 threads = 8 waves (2M x 4N), each wave one 16x16 C tile,
// K staged in chunks of 64 via LDS (2 WMMA per stage).
// ---------------------------------------------------------------------------
template <int BMODE, int EPMODE>
__global__ __launch_bounds__(256) void gemm_wmma_kernel(
    const half_t* __restrict__ A, const half_t* __restrict__ act,
    const float* __restrict__ bias, float* __restrict__ out,
    int M, int N, int K, int Cin, int lgW, int upD, int upK) {
  __shared__ __align__(16) half_t As[BM][KPAD];
  __shared__ __align__(16) half_t Bs[BN][KPAD];  // K-transposed: Bs[n][k]

  const int tid = threadIdx.x;
  const int lane = tid & 31;
  const int wave = tid >> 5;
  const int wm = wave >> 2;  // 0..1
  const int wn = wave & 3;   // 0..3
  const int tileN0 = blockIdx.x * BN;
  const int tileM0 = blockIdx.y * BM;
  const int Wd = 1 << lgW;
  const int lgHW = 2 * lgW;
  const int HWm1 = (1 << lgHW) - 1;
  const bool kAligned8 = (K & 7) == 0;
#if HAVE_ASYNC_LDS
  const bool useAsync = (K & (BK - 1)) == 0;  // full-chunk rows only
#endif

  v8f acc = {0.f, 0.f, 0.f, 0.f, 0.f, 0.f, 0.f, 0.f};

  for (int k0 = 0; k0 < K; k0 += BK) {
    if (tid == 0 && k0 + BK < K)  // global_prefetch_b8 hint for next A panel
      __builtin_prefetch(&A[(size_t)tileM0 * K + k0 + BK], 0, 1);

    // ---- stage A tile (32 x 64 halves): one 16B chunk per thread ----
    {
      const int mm = tid >> 3, j = tid & 7;
      const int gm = tileM0 + mm;
      const int gk0 = k0 + j * 8;
#if HAVE_ASYNC_LDS
      if (useAsync) {
        // rows beyond M read adjacent ws data; masked in the epilogue.
        __builtin_amdgcn_global_load_async_to_lds_b128(
            (__attribute__((address_space(1))) v4i*)(A + (size_t)gm * K + gk0),
            (__attribute__((address_space(3))) v4i*)&As[mm][j * 8], 0, 0);
      } else
#endif
      {
        union { v8h v; half_t h[8]; } t;
        if (kAligned8 && gm < M && gk0 + 8 <= K) {
          t.v = *(const v8h*)(A + (size_t)gm * K + gk0);
        } else {
#pragma unroll
          for (int i = 0; i < 8; ++i) {
            int gk = gk0 + i;
            int gkc = gk < K ? gk : K - 1;           // clamped, branchless
            half_t v = A[(size_t)(gm < M ? gm : 0) * K + gkc];
            t.h[i] = (gm < M && gk < K) ? v : (half_t)0.f;
          }
        }
        *(v8h*)&As[mm][j * 8] = t.v;
      }
    }

    // ---- stage B tile (64 x 64 halves), implicit im2col, branchless ----
#pragma unroll
    for (int o = tid; o < (BN * BK) / 8; o += 256) {  // 512 octets
      const int nn = o >> 3, j = o & 7;
      const int gn = tileN0 + nn;  // N is always a multiple of BN
      const int b = gn >> lgHW, hw = gn & HWm1;
      union { v8h v; half_t h[8]; } t;
      if (BMODE == BMODE_CONV1) {
#pragma unroll
        for (int i = 0; i < 8; ++i) {
          int gk = k0 + j * 8 + i;
          int gkc = gk < K ? gk : K - 1;
          half_t v = act[((size_t)(b * Cin + gkc) << lgHW) + hw];
          t.h[i] = (gk < K) ? v : (half_t)0.f;
        }
      } else {
        const int y = hw >> lgW, x = hw & (Wd - 1);
#pragma unroll
        for (int i = 0; i < 8; ++i) {
          int gk = k0 + j * 8 + i;
          int gkc = gk < K ? gk : K - 1;
          int ci = gkc / 9, r9 = gkc - ci * 9;
          int dy = r9 / 3 - 1, dx = r9 - (r9 / 3) * 3 - 1;
          int yy = y + dy, xx = x + dx;
          bool ok = (gk < K) & (yy >= 0) & (yy < Wd) & (xx >= 0) & (xx < Wd);
          int yyc = yy < 0 ? 0 : (yy >= Wd ? Wd - 1 : yy);
          int xxc = xx < 0 ? 0 : (xx >= Wd ? Wd - 1 : xx);
          half_t v = act[((size_t)(b * Cin + ci) << lgHW) + (yyc << lgW) + xxc];
          t.h[i] = ok ? v : (half_t)0.f;
        }
      }
      *(v8h*)&Bs[nn][j * 8] = t.v;
    }
#if HAVE_ASYNC_LDS
    if (useAsync) wait_async_lds0();
#endif
    __syncthreads();

    // ---- 2x WMMA per stage; 16-bit A/B lane layout:
    // lane&15 -> row/col, lane>=16 -> K halves {8..15, 24..31} of sub-chunk.
    const int l15 = lane & 15;
    const int kb = (lane >> 4) * 8;
    const half_t* arow = &As[wm * 16 + l15][0];
    const half_t* brow = &Bs[wn * 16 + l15][0];
#pragma unroll
    for (int ks = 0; ks < 2; ++ks) {
      const int o32 = ks * 32;
      union { v16h v; half_t h[16]; } af, bf;
#pragma unroll
      for (int i = 0; i < 8; ++i) {
        af.h[i] = arow[o32 + kb + i];
        af.h[8 + i] = arow[o32 + kb + 16 + i];
        bf.h[i] = brow[o32 + kb + i];
        bf.h[8 + i] = brow[o32 + kb + 16 + i];
      }
      acc = __builtin_amdgcn_wmma_f32_16x16x32_f16(
          false, af.v, false, bf.v, (short)0, acc, false, false);
    }
    __syncthreads();
  }

  // ---- epilogue: C/D layout -> lane&15 = col, rows ((lane>=16)?8:0)+i ----
  const int col = tileN0 + wn * 16 + (lane & 15);
  const int mbase = tileM0 + wm * 16 + ((lane >> 4) * 8);
  const int b = col >> lgHW, hw = col & HWm1;
#pragma unroll
  for (int i = 0; i < 8; ++i) {
    int m = mbase + i;
    if (m < M) {
      float v = acc[i] + (bias ? bias[m] : 0.f);
      if (EPMODE == EP_PLAIN) {
        out[((size_t)(b * M + m) << lgHW) + hw] = v;
      } else {
        int y = hw >> lgW, x = hw & (Wd - 1);
        out[((size_t)(b * M + m) << (lgHW + 2)) +
            ((size_t)((2 * y + upD)) << (lgW + 1)) + 2 * x + upK] = v;
      }
    }
  }
}

// ---------------------------------------------------------------------------
// Instance norm (+ optional residual add) + optional exact GELU, in place.
// One block per (b*C + c) plane.
// ---------------------------------------------------------------------------
__global__ __launch_bounds__(256) void instnorm_kernel(
    float* __restrict__ x, const float* __restrict__ addsrc, int HW,
    int do_gelu) {
  __shared__ float s_sum[256], s_sq[256];
  const size_t base = (size_t)blockIdx.x * HW;
  float sum = 0.f, sq = 0.f;
  for (int i = threadIdx.x; i < HW; i += 256) {
    float v = x[base + i];
    if (addsrc) v += addsrc[base + i];
    sum += v;
    sq += v * v;
  }
  s_sum[threadIdx.x] = sum;
  s_sq[threadIdx.x] = sq;
  __syncthreads();
  for (int s = 128; s > 0; s >>= 1) {
    if (threadIdx.x < s) {
      s_sum[threadIdx.x] += s_sum[threadIdx.x + s];
      s_sq[threadIdx.x] += s_sq[threadIdx.x + s];
    }
    __syncthreads();
  }
  const float mu = s_sum[0] / (float)HW;
  const float var = s_sq[0] / (float)HW - mu * mu;
  const float rs = rsqrtf(var + 1e-5f);
  for (int i = threadIdx.x; i < HW; i += 256) {
    float v = x[base + i];
    if (addsrc) v += addsrc[base + i];
    v = (v - mu) * rs;
    if (do_gelu) v = 0.5f * v * (1.f + erff(v * 0.70710678118654752f));
    x[base + i] = v;
  }
}

// f32 -> f16 with channel placement (all dims power-of-two, shift addressing)
__global__ void cvt_f16_kernel(const float* __restrict__ src,
                               half_t* __restrict__ dst, int lgC, int lgHW,
                               int lgDstC, int c0, int total) {
  int idx = blockIdx.x * 256 + threadIdx.x;
  if (idx >= total) return;
  int i = idx & ((1 << lgHW) - 1);
  int c = (idx >> lgHW) & ((1 << lgC) - 1);
  int b = idx >> (lgHW + lgC);
  dst[((size_t)((b << lgDstC) + c0 + c) << lgHW) + i] = (half_t)src[idx];
}

__global__ void copy_f16_kernel(const half_t* __restrict__ src,
                                half_t* __restrict__ dst, int lgC, int lgHW,
                                int lgDstC, int c0, int total) {
  int idx = blockIdx.x * 256 + threadIdx.x;
  if (idx >= total) return;
  int i = idx & ((1 << lgHW) - 1);
  int c = (idx >> lgHW) & ((1 << lgC) - 1);
  int b = idx >> (lgHW + lgC);
  dst[((size_t)((b << lgDstC) + c0 + c) << lgHW) + i] = src[idx];
}

__global__ void cvt_flat_kernel(const float* __restrict__ s,
                                half_t* __restrict__ d, int n) {
  int idx = blockIdx.x * 256 + threadIdx.x;
  if (idx < n) d[idx] = (half_t)s[idx];
}

// 2x2 maxpool on f16 NCHW (square power-of-two input)
__global__ void maxpool_f16_kernel(const half_t* __restrict__ src,
                                   half_t* __restrict__ dst, int lgW,
                                   int total) {
  int idx = blockIdx.x * 256 + threadIdx.x;
  if (idx >= total) return;
  int lgOW = lgW - 1;
  int ow = idx & ((1 << lgOW) - 1);
  int oh = (idx >> lgOW) & ((1 << lgOW) - 1);
  int plane = idx >> (2 * lgOW);
  const half_t* p =
      src + ((size_t)plane << (2 * lgW)) + ((size_t)(2 * oh) << lgW) + 2 * ow;
  float a = (float)p[0], b = (float)p[1];
  float c = (float)p[1 << lgW], d = (float)p[(1 << lgW) + 1];
  dst[idx] = (half_t)fmaxf(fmaxf(a, b), fmaxf(c, d));
}

// conv-transpose weight repack: w[c,o,d,k](f32) -> out[(d*2+k), o, c](f16)
__global__ void repack_ct_kernel(const float* __restrict__ w,
                                 half_t* __restrict__ out, int Cin, int O,
                                 int total) {
  int idx = blockIdx.x * 256 + threadIdx.x;
  if (idx >= total) return;
  int k = idx & 1;
  int d = (idx >> 1) & 1;
  int t = idx >> 2;
  int o = t & (O - 1);
  int c = t / O;
  out[(size_t)((d * 2 + k) * O + o) * Cin + c] =
      (half_t)w[((size_t)(c * O + o) * 2 + d) * 2 + k];
}

// ---------------------------------------------------------------------------
// Spectral conv (H=W=64, 16x16 modes kept on rows {0..15, 48..63}).
// TB layout (f32): FWc[0]=64x16, FWs[1024], FHc[2048]=64x32, FHs[4096],
// IHc[6144]=32x64, IHs[8192], IWc[10240]=16x64, IWs[11264].
// ---------------------------------------------------------------------------
__global__ void spectral_tables_kernel(float* __restrict__ tb) {
  const float PI2 = 6.283185307179586f;
  int stride = gridDim.x * 256;
  for (int i = blockIdx.x * 256 + threadIdx.x; i < 1024; i += stride) {
    int w = i >> 4, ky = i & 15;
    float a = -PI2 * (float)(w * ky) / 64.f;
    tb[i] = cosf(a);
    tb[1024 + i] = sinf(a);
  }
  for (int i = blockIdx.x * 256 + threadIdx.x; i < 2048; i += stride) {
    int h = i >> 5, kxi = i & 31;
    int kx = kxi < 16 ? kxi : kxi + 32;
    float a = -PI2 * (float)(h * kx) / 64.f;
    tb[2048 + i] = cosf(a);
    tb[4096 + i] = sinf(a);
  }
  for (int i = blockIdx.x * 256 + threadIdx.x; i < 2048; i += stride) {
    int kxi = i >> 6, h = i & 63;
    int kx = kxi < 16 ? kxi : kxi + 32;
    float a = PI2 * (float)(kx * h) / 64.f;
    tb[6144 + i] = cosf(a);
    tb[8192 + i] = sinf(a);
  }
  for (int i = blockIdx.x * 256 + threadIdx.x; i < 1024; i += stride) {
    int ky = i >> 6, w = i & 63;
    float a = PI2 * (float)(ky * w) / 64.f;
    float s = (ky == 0 ? 1.f : 2.f) / 4096.f;  // irfft weights + 1/(H*W)
    tb[10240 + i] = cosf(a) * s;
    tb[11264 + i] = sinf(a) * s;
  }
}

// T1[b,c,h,ky] (complex) = sum_w x[b,c,h,w] * e^{-i2pi w ky/64}
__global__ void fno_fwd_w_kernel(const float* __restrict__ x,
                                 const float* __restrict__ tb,
                                 float* __restrict__ T1, int total) {
  int idx = blockIdx.x * 256 + threadIdx.x;
  if (idx >= total) return;
  int ky = idx & 15, h = (idx >> 4) & 63, bc = idx >> 10;
  const float* row = x + ((size_t)bc * 64 + h) * 64;
  float re = 0.f, im = 0.f;
  for (int w = 0; w < 64; ++w) {
    float v = row[w];
    re += v * tb[w * 16 + ky];
    im += v * tb[1024 + w * 16 + ky];
  }
  T1[(size_t)idx * 2] = re;
  T1[(size_t)idx * 2 + 1] = im;
}

// XF[b,c,kxi,ky] = sum_h T1[b,c,h,ky] * e^{-i2pi h kx/64}
__global__ void fno_fwd_h_kernel(const float* __restrict__ T1,
                                 const float* __restrict__ tb,
                                 float* __restrict__ XF, int total) {
  int idx = blockIdx.x * 256 + threadIdx.x;
  if (idx >= total) return;
  int ky = idx & 15, kxi = (idx >> 4) & 31, bc = idx >> 9;
  float re = 0.f, im = 0.f;
  for (int h = 0; h < 64; ++h) {
    float tr = T1[((size_t)(bc * 64 + h) * 16 + ky) * 2];
    float ti = T1[((size_t)(bc * 64 + h) * 16 + ky) * 2 + 1];
    float fc = tb[2048 + h * 32 + kxi];
    float fs = tb[4096 + h * 32 + kxi];
    re += tr * fc - ti * fs;
    im += tr * fs + ti * fc;
  }
  XF[(size_t)idx * 2] = re;
  XF[(size_t)idx * 2 + 1] = im;
}

// OFT[b,o,kxi,ky] = sum_i XF[b,i,kxi,ky] * W[i,o,kx,ky]   (complex)
__global__ void fno_mix_kernel(const float* __restrict__ XF,
                               const float* __restrict__ w1r,
                               const float* __restrict__ w1i,
                               const float* __restrict__ w2r,
                               const float* __restrict__ w2i,
                               float* __restrict__ OFT, int total) {
  int idx = blockIdx.x * 256 + threadIdx.x;
  if (idx >= total) return;
  int ky = idx & 15, kxi = (idx >> 4) & 31, o = (idx >> 9) & 63, b = idx >> 15;
  const float* wr = (kxi < 16) ? w1r : w2r;
  const float* wi = (kxi < 16) ? w1i : w2i;
  int kxl = kxi & 15;
  float re = 0.f, im = 0.f;
  for (int i = 0; i < 64; ++i) {
    size_t xo = (((size_t)(b * 64 + i) * 32 + kxi) * 16 + ky) * 2;
    float xr = XF[xo], xi = XF[xo + 1];
    size_t wo = ((size_t)(i * 64 + o) * 16 + kxl) * 16 + ky;
    float wrv = wr[wo], wiv = wi[wo];
    re += xr * wrv - xi * wiv;
    im += xr * wiv + xi * wrv;
  }
  size_t oo = (((size_t)(b * 64 + o) * 32 + kxi) * 16 + ky) * 2;
  OFT[oo] = re;
  OFT[oo + 1] = im;
}

// T2[b,o,h,ky] = sum_kxi OFT[b,o,kxi,ky] * e^{+i2pi kx h/64}
__global__ void fno_inv_h_kernel(const float* __restrict__ OFT,
                                 const float* __restrict__ tb,
                                 float* __restrict__ T2, int total) {
  int idx = blockIdx.x * 256 + threadIdx.x;
  if (idx >= total) return;
  int ky = idx & 15, h = (idx >> 4) & 63, bo = idx >> 10;
  float re = 0.f, im = 0.f;
  for (int kxi = 0; kxi < 32; ++kxi) {
    size_t fo = ((size_t)(bo * 32 + kxi) * 16 + ky) * 2;
    float fr = OFT[fo], fi = OFT[fo + 1];
    float ic = tb[6144 + kxi * 64 + h];
    float is = tb[8192 + kxi * 64 + h];
    re += fr * ic - fi * is;
    im += fr * is + fi * ic;
  }
  T2[(size_t)idx * 2] = re;
  T2[(size_t)idx * 2 + 1] = im;
}

// Y[b,o,h,w] = sum_ky Re(T2[b,o,h,ky] * e^{+i2pi ky w/64}) * weight/(H*W)
__global__ void fno_inv_w_kernel(const float* __restrict__ T2,
                                 const float* __restrict__ tb,
                                 float* __restrict__ Y, int total) {
  int idx = blockIdx.x * 256 + threadIdx.x;
  if (idx >= total) return;
  int w = idx & 63, h = (idx >> 6) & 63, bo = idx >> 12;
  float acc = 0.f;
  for (int ky = 0; ky < 16; ++ky) {
    size_t to = ((size_t)(bo * 64 + h) * 16 + ky) * 2;
    acc += T2[to] * tb[10240 + ky * 64 + w] -
           T2[to + 1] * tb[11264 + ky * 64 + w];
  }
  Y[idx] = acc;
}

// ---------------------------------------------------------------------------
// Host orchestration
// ---------------------------------------------------------------------------
static void cvt16(const float* s, half_t* d, int C, int HW, int dstC, int c0,
                  int B, hipStream_t st) {
  int total = B * C * HW;
  cvt_f16_kernel<<<ceil_div(total, 256), 256, 0, st>>>(
      s, d, ilog2i(C), ilog2i(HW), ilog2i(dstC), c0, total);
}
static void cvt16_flat(const float* s, half_t* d, int n, hipStream_t st) {
  cvt_flat_kernel<<<ceil_div(n, 256), 256, 0, st>>>(s, d, n);
}
static void cpy16(const half_t* s, half_t* d, int C, int HW, int dstC, int c0,
                  int B, hipStream_t st) {
  int total = B * C * HW;
  copy_f16_kernel<<<ceil_div(total, 256), 256, 0, st>>>(
      s, d, ilog2i(C), ilog2i(HW), ilog2i(dstC), c0, total);
}
static void gemm3(const half_t* A, const half_t* act, const float* bias,
                  float* out, int M, int N, int K, int Cin, int lgW,
                  hipStream_t st) {
  dim3 g(ceil_div(N, BN), ceil_div(M, BM));
  gemm_wmma_kernel<BMODE_CONV3, EP_PLAIN>
      <<<g, 256, 0, st>>>(A, act, bias, out, M, N, K, Cin, lgW, 0, 0);
}
static void gemm1(const half_t* A, const half_t* act, const float* bias,
                  float* out, int M, int N, int K, int lgW, hipStream_t st) {
  dim3 g(ceil_div(N, BN), ceil_div(M, BM));
  gemm_wmma_kernel<BMODE_CONV1, EP_PLAIN>
      <<<g, 256, 0, st>>>(A, act, bias, out, M, N, K, K, lgW, 0, 0);
}
static void gemm_up(const half_t* A, const half_t* act, const float* bias,
                    float* out, int M, int N, int K, int lgW, int d, int k,
                    hipStream_t st) {
  dim3 g(ceil_div(N, BN), ceil_div(M, BM));
  gemm_wmma_kernel<BMODE_CONV1, EP_UP2>
      <<<g, 256, 0, st>>>(A, act, bias, out, M, N, K, K, lgW, d, k);
}
static void norm(float* x, const float* add, int BC, int HW, int gelu,
                 hipStream_t st) {
  instnorm_kernel<<<BC, 256, 0, st>>>(x, add, HW, gelu);
}

extern "C" void kernel_launch(void* const* d_in, const int* in_sizes, int n_in,
                              void* d_out, int out_size, void* d_ws,
                              size_t ws_size, hipStream_t stream) {
  (void)in_sizes; (void)n_in; (void)out_size; (void)ws_size;
  // ---- input pytree (jax tree-flatten: top-level dict order x, params;
  //      nested dict keys sorted alphabetically, lists in order) ----
  int ii = 0;
  const float* x_in = (const float*)d_in[ii++];
  const float *db_b1[3], *db_b2[3], *db_w1[3], *db_w2[3];
  for (int l = 0; l < 3; ++l) {
    db_b1[l] = (const float*)d_in[ii++];
    db_b2[l] = (const float*)d_in[ii++];
    db_w1[l] = (const float*)d_in[ii++];
    db_w2[l] = (const float*)d_in[ii++];
  }
  const float *du_b[3], *du_w[3];
  for (int l = 0; l < 3; ++l) {
    du_b[l] = (const float*)d_in[ii++];
    du_w[l] = (const float*)d_in[ii++];
  }
  const float *en_b1[3], *en_b2[3], *en_w1[3], *en_w2[3];
  for (int l = 0; l < 3; ++l) {
    en_b1[l] = (const float*)d_in[ii++];
    en_b2[l] = (const float*)d_in[ii++];
    en_w1[l] = (const float*)d_in[ii++];
    en_w2[l] = (const float*)d_in[ii++];
  }
  const float* fin_b = (const float*)d_in[ii++];
  const float* fin_w = (const float*)d_in[ii++];
  const float *fno_cb[4], *fno_cw[4], *fno_w1i[4], *fno_w1r[4], *fno_w2i[4],
      *fno_w2r[4];
  for (int p = 0; p < 4; ++p) {
    fno_cb[p] = (const float*)d_in[ii++];
    fno_cw[p] = (const float*)d_in[ii++];
    fno_w1i[p] = (const float*)d_in[ii++];
    fno_w1r[p] = (const float*)d_in[ii++];
    fno_w2i[p] = (const float*)d_in[ii++];
    fno_w2r[p] = (const float*)d_in[ii++];
  }
  const float* pin_b = (const float*)d_in[ii++];
  const float* pin_w = (const float*)d_in[ii++];
  const float* pout_b = (const float*)d_in[ii++];
  const float* pout_w = (const float*)d_in[ii++];

  // ---- workspace bump allocator ----
  uint8_t* wsb = (uint8_t*)d_ws;
  size_t off = 0;
  auto alloc = [&](size_t bytes) -> void* {
    void* p = wsb + off;
    off += (bytes + 255) & ~(size_t)255;
    return p;
  };
  const int B = 2, HW0 = 512 * 512;
  half_t* PA = (half_t*)alloc((size_t)64 * B * HW0 * 2);  // 67 MB
  half_t* PB = (half_t*)alloc((size_t)64 * B * HW0 * 2);  // 67 MB
  float*  FA = (float*)alloc((size_t)32 * B * HW0 * 4);   // 67 MB
  half_t* SK0 = (half_t*)alloc((size_t)32 * B * HW0 * 2);
  half_t* SK1 = (half_t*)alloc((size_t)64 * B * 65536 * 2);
  half_t* SK2 = (half_t*)alloc((size_t)128 * B * 16384 * 2);
  float* FH32 = (float*)alloc((size_t)64 * B * 4096 * 4);
  float* TB = (float*)alloc(12288 * 4);
  float* T1 = (float*)alloc(262144 * 4);
  float* XF = (float*)alloc(131072 * 4);
  float* OFT = (float*)alloc(131072 * 4);
  float* T2 = (float*)alloc(262144 * 4);
  float* Y = (float*)alloc(524288 * 4);
  // f16 weights
  half_t* e16w1[3];
  half_t* e16w2[3];
  int enc_ch[3] = {32, 64, 128}, enc_prev[3] = {2, 32, 64};
  for (int l = 0; l < 3; ++l) {
    e16w1[l] = (half_t*)alloc((size_t)enc_ch[l] * enc_prev[l] * 9 * 2);
    e16w2[l] = (half_t*)alloc((size_t)enc_ch[l] * enc_ch[l] * 9 * 2);
  }
  half_t* d16w1[3];
  half_t* d16w2[3];
  half_t* ct16[3];
  int dec_ch[3] = {128, 64, 32}, ct_in[3] = {128, 128, 64};
  for (int l = 0; l < 3; ++l) {
    d16w1[l] = (half_t*)alloc((size_t)dec_ch[l] * 2 * dec_ch[l] * 9 * 2);
    d16w2[l] = (half_t*)alloc((size_t)dec_ch[l] * dec_ch[l] * 9 * 2);
    ct16[l] = (half_t*)alloc((size_t)4 * dec_ch[l] * ct_in[l] * 2);
  }
  half_t* pin16 = (half_t*)alloc(64 * 128 * 2);
  half_t* pout16 = (half_t*)alloc(128 * 64 * 2);
  half_t* cw16[4];
  for (int p = 0; p < 4; ++p) cw16[p] = (half_t*)alloc(64 * 64 * 2);
  half_t* fin16 = (half_t*)alloc(2 * 32 * 2);
  (void)alloc(1 << 20);  // tail slack (A-panel over-reads past last weight)

  // ---- weight conversion / repack ----
  for (int l = 0; l < 3; ++l) {
    cvt16_flat(en_w1[l], e16w1[l], enc_ch[l] * enc_prev[l] * 9, stream);
    cvt16_flat(en_w2[l], e16w2[l], enc_ch[l] * enc_ch[l] * 9, stream);
    cvt16_flat(db_w1[l], d16w1[l], dec_ch[l] * 2 * dec_ch[l] * 9, stream);
    cvt16_flat(db_w2[l], d16w2[l], dec_ch[l] * dec_ch[l] * 9, stream);
    int tot = ct_in[l] * dec_ch[l] * 4;
    repack_ct_kernel<<<ceil_div(tot, 256), 256, 0, stream>>>(
        du_w[l], ct16[l], ct_in[l], dec_ch[l], tot);
  }
  cvt16_flat(pin_w, pin16, 64 * 128, stream);
  cvt16_flat(pout_w, pout16, 128 * 64, stream);
  for (int p = 0; p < 4; ++p) cvt16_flat(fno_cw[p], cw16[p], 64 * 64, stream);
  cvt16_flat(fin_w, fin16, 2 * 32, stream);
  spectral_tables_kernel<<<8, 256, 0, stream>>>(TB);

  // ---- encoder ----
  cvt16(x_in, PA, 2, HW0, 2, 0, B, stream);
  // enc0 @512 (lgW=9)
  gemm3(e16w1[0], PA, en_b1[0], FA, 32, B * HW0, 18, 2, 9, stream);
  norm(FA, nullptr, B * 32, HW0, 1, stream);
  cvt16(FA, PB, 32, HW0, 32, 0, B, stream);
  gemm3(e16w2[0], PB, en_b2[0], FA, 32, B * HW0, 288, 32, 9, stream);
  norm(FA, nullptr, B * 32, HW0, 1, stream);
  cvt16(FA, SK0, 32, HW0, 32, 0, B, stream);
  maxpool_f16_kernel<<<ceil_div(B * 32 * 65536, 256), 256, 0, stream>>>(
      SK0, PA, 9, B * 32 * 65536);
  // enc1 @256 (lgW=8)
  gemm3(e16w1[1], PA, en_b1[1], FA, 64, B * 65536, 288, 32, 8, stream);
  norm(FA, nullptr, B * 64, 65536, 1, stream);
  cvt16(FA, PB, 64, 65536, 64, 0, B, stream);
  gemm3(e16w2[1], PB, en_b2[1], FA, 64, B * 65536, 576, 64, 8, stream);
  norm(FA, nullptr, B * 64, 65536, 1, stream);
  cvt16(FA, SK1, 64, 65536, 64, 0, B, stream);
  maxpool_f16_kernel<<<ceil_div(B * 64 * 16384, 256), 256, 0, stream>>>(
      SK1, PA, 8, B * 64 * 16384);
  // enc2 @128 (lgW=7)
  gemm3(e16w1[2], PA, en_b1[2], FA, 128, B * 16384, 576, 64, 7, stream);
  norm(FA, nullptr, B * 128, 16384, 1, stream);
  cvt16(FA, PB, 128, 16384, 128, 0, B, stream);
  gemm3(e16w2[2], PB, en_b2[2], FA, 128, B * 16384, 1152, 128, 7, stream);
  norm(FA, nullptr, B * 128, 16384, 1, stream);
  cvt16(FA, SK2, 128, 16384, 128, 0, B, stream);
  maxpool_f16_kernel<<<ceil_div(B * 128 * 4096, 256), 256, 0, stream>>>(
      SK2, PA, 7, B * 128 * 4096);

  // ---- proj_in @64 (lgW=6) ----
  gemm1(pin16, PA, pin_b, FA, 64, B * 4096, 128, 6, stream);
  (void)hipMemcpyAsync(FH32, FA, (size_t)64 * B * 4096 * 4,
                       hipMemcpyDeviceToDevice, stream);
  cvt16(FA, PA, 64, 4096, 64, 0, B, stream);

  // ---- 4x FNO blocks @64 ----
  for (int p = 0; p < 4; ++p) {
    fno_fwd_w_kernel<<<ceil_div(131072, 256), 256, 0, stream>>>(FH32, TB, T1,
                                                                131072);
    fno_fwd_h_kernel<<<ceil_div(65536, 256), 256, 0, stream>>>(T1, TB, XF,
                                                               65536);
    fno_mix_kernel<<<ceil_div(65536, 256), 256, 0, stream>>>(
        XF, fno_w1r[p], fno_w1i[p], fno_w2r[p], fno_w2i[p], OFT, 65536);
    fno_inv_h_kernel<<<ceil_div(131072, 256), 256, 0, stream>>>(OFT, TB, T2,
                                                                131072);
    fno_inv_w_kernel<<<ceil_div(524288, 256), 256, 0, stream>>>(T2, TB, Y,
                                                                524288);
    gemm1(cw16[p], PA, fno_cb[p], FA, 64, B * 4096, 64, 6, stream);
    norm(FA, Y, B * 64, 4096, 1, stream);
    (void)hipMemcpyAsync(FH32, FA, (size_t)64 * B * 4096 * 4,
                         hipMemcpyDeviceToDevice, stream);
    cvt16(FA, PA, 64, 4096, 64, 0, B, stream);
  }

  // ---- proj_out @64 ----
  gemm1(pout16, PA, pout_b, FA, 128, B * 4096, 64, 6, stream);
  cvt16(FA, PA, 128, 4096, 128, 0, B, stream);

  // ---- decoder ----
  half_t* skips[3] = {SK2, SK1, SK0};
  int inLgW[3] = {6, 7, 8};
  for (int l = 0; l < 3; ++l) {
    int ch = dec_ch[l], inc = ct_in[l];
    int lw = inLgW[l];
    int Nin = B << (2 * lw);
    int HWo = 1 << (2 * (lw + 1));
    for (int dk = 0; dk < 4; ++dk)
      gemm_up(ct16[l] + (size_t)dk * ch * inc, PA, du_b[l], FA, ch, Nin, inc,
              lw, dk >> 1, dk & 1, stream);
    cvt16(FA, PB, ch, HWo, 2 * ch, 0, B, stream);
    cpy16(skips[l], PB, ch, HWo, 2 * ch, ch, B, stream);
    gemm3(d16w1[l], PB, db_b1[l], FA, ch, B * HWo, 2 * ch * 9, 2 * ch, lw + 1,
          stream);
    norm(FA, nullptr, B * ch, HWo, 1, stream);
    cvt16(FA, PA, ch, HWo, ch, 0, B, stream);
    gemm3(d16w2[l], PA, db_b2[l], FA, ch, B * HWo, ch * 9, ch, lw + 1, stream);
    norm(FA, nullptr, B * ch, HWo, 1, stream);
    cvt16(FA, PA, ch, HWo, ch, 0, B, stream);
  }

  // ---- final 1x1 conv: 32 -> 2 @512, straight to d_out (f32) ----
  gemm1(fin16, PA, fin_b, (float*)d_out, 2, B * HW0, 32, 9, stream);
}
